// LGRLSeqRefineDecoder_10290741641693
// MI455X (gfx1250) — matrix operand order repaired and checked by reading, not compiled, not run on hardware
//
#include <hip/hip_runtime.h>
#include <hip/hip_bf16.h>
#include <cstdint>

// ---------------- problem constants ----------------
static constexpr int kB  = 128;
static constexpr int kP  = 5;
static constexpr int kT  = 128;
static constexpr int kLC = 64;
static constexpr int kLT = 256;
static constexpr int kV  = 2048;
static constexpr int kE  = 256;
static constexpr int kH  = 256;
static constexpr int kM  = 512;
static constexpr int kN  = kB * kP;          // 640 rows
static constexpr int kIN = kE + kM + kM;     // 1280
static constexpr int kXH0 = kIN + kH;        // 1536 : [tok | io | ctx | h0]
static constexpr int kXH1 = 2 * kH;          // 512  : [h0 | h1]
static constexpr float kSCALE = 0.044194173824159216f; // 1/sqrt(512)

// ---------------- WMMA types ----------------
typedef __attribute__((ext_vector_type(16))) __bf16 v16bf;
typedef __attribute__((ext_vector_type(8)))  float  v8f;

union Frag32B { uint4 u[2]; v16bf v; };

__device__ __forceinline__ float sigmoidf_(float x) { return 1.0f / (1.0f + __expf(-x)); }
__device__ __forceinline__ float bflo_(unsigned u) { return __uint_as_float(u << 16); }
__device__ __forceinline__ float bfhi_(unsigned u) { return __uint_as_float(u & 0xffff0000u); }

// ---------------- elementwise / packing kernels ----------------
__global__ void k_zero32(unsigned* p, long long n) {
  long long i = (long long)blockIdx.x * blockDim.x + threadIdx.x;
  if (i < n) p[i] = 0u;
}

__global__ void k_cvt_bf16(__bf16* dst, const float* src, long long n) {
  long long i = (long long)blockIdx.x * blockDim.x + threadIdx.x;
  if (i < n) dst[i] = (__bf16)src[i];
}

__global__ void k_copy_f32(float* dst, const float* src, long long n) {
  long long i = (long long)blockIdx.x * blockDim.x + threadIdx.x;
  if (i < n) dst[i] = src[i];
}

__global__ void k_add_f32(float* dst, const float* a, const float* b, long long n) {
  long long i = (long long)blockIdx.x * blockDim.x + threadIdx.x;
  if (i < n) dst[i] = a[i] + b[i];
}

// pack [A | B] along K into bf16: dst is rows x (Ka+Kb)
__global__ void k_pack2_bf16(__bf16* dst, const float* a, int Ka,
                             const float* b, int Kb, long long rows) {
  long long i = (long long)blockIdx.x * blockDim.x + threadIdx.x;
  int Kt = Ka + Kb;
  if (i >= rows * (long long)Kt) return;
  long long r = i / Kt;
  int k = (int)(i % Kt);
  float v = (k < Ka) ? a[r * Ka + k] : b[r * Kb + (k - Ka)];
  dst[i] = (__bf16)v;
}

// write io_embed bf16 into xh0 slot [256, 768)
__global__ void k_fill_io(__bf16* xh0, const float* io) {
  long long i = (long long)blockIdx.x * blockDim.x + threadIdx.x;
  if (i >= (long long)kN * kM) return;
  int n = (int)(i >> 9), d = (int)(i & 511);
  xh0[(long long)n * kXH0 + kE + d] = (__bf16)io[i];
}

// gather token embedding into xh0 slot [0, 256)
__global__ void k_tok(__bf16* xh0, const __bf16* embedB, const long long* outs, int t) {
  long long i = (long long)blockIdx.x * blockDim.x + threadIdx.x;
  if (i >= (long long)kN * kE) return;
  int n = (int)(i >> 8), e = (int)(i & 255);
  int b = n / kP;
  long long tok = outs[(long long)b * (kT + 1) + t];
  xh0[(long long)n * kXH0 + e] = embedB[tok * kE + e];
}

// LSTM activation: gates (N x 1024 f32, [i|f|g|o]), c updated in place,
// h written as bf16 to two strided destinations.
__global__ void k_lstm_act(const float* __restrict__ gates, float* __restrict__ c,
                           __bf16* __restrict__ h1, int s1,
                           __bf16* __restrict__ h2, int s2) {
  long long i = (long long)blockIdx.x * blockDim.x + threadIdx.x;
  if (i >= (long long)kN * kH) return;
  int n = (int)(i >> 8), j = (int)(i & 255);
  const float* g = gates + (long long)n * (4 * kH);
  float gi = sigmoidf_(g[j]);
  float gf = sigmoidf_(g[j + kH]);
  float gg = tanhf(g[j + 2 * kH]);
  float go = sigmoidf_(g[j + 3 * kH]);
  float cn = gf * c[i] + gi * gg;
  c[i] = cn;
  float h = go * tanhf(cn);
  h1[(long long)n * s1 + j] = (__bf16)h;
  h2[(long long)n * s2 + j] = (__bf16)h;
}

// feat[b, j] = max_p concat(ctx, dec)[b*P+p, j]; output bf16 (128 x 768)
__global__ void k_maxfeat(const float* __restrict__ ctx, const __bf16* __restrict__ dec,
                          __bf16* __restrict__ feat) {
  long long i = (long long)blockIdx.x * blockDim.x + threadIdx.x;
  if (i >= (long long)kB * (kM + kH)) return;
  int b = (int)(i / (kM + kH)), j = (int)(i % (kM + kH));
  float m = -3.0e38f;
#pragma unroll
  for (int p = 0; p < kP; ++p) {
    int n = b * kP + p;
    float v = (j < kM) ? ctx[(long long)n * kM + j]
                       : (float)dec[(long long)n * kH + (j - kM)];
    m = fmaxf(m, v);
  }
  feat[i] = (__bf16)m;
}

// labels tail: float cast of outputs[:, 1:]
__global__ void k_labels(float* out, const long long* outs) {
  long long i = (long long)blockIdx.x * blockDim.x + threadIdx.x;
  if (i >= (long long)kB * kT) return;
  int b = (int)(i / kT), t = (int)(i % kT);
  out[(long long)kB * kT * kV + i] = (float)outs[(long long)b * (kT + 1) + t + 1];
}

// ---------------- WMMA GEMM ----------------
// C[M x N] = A[M x K](bf16, lda=K) * W[N x K]^T(bf16) + bias[N]
// Each wave computes a 16(M) x 64(N) tile with v_wmma_f32_16x16x32_bf16.
// K-loop is double-buffered (all K are multiples of 64) so the 10 fragment
// loads of block k+32 are in flight while the 4 WMMAs of block k execute.
__global__ __launch_bounds__(256) void k_gemm_wmma(
    const __bf16* __restrict__ A, const __bf16* __restrict__ W,
    const float* __restrict__ bias,
    float* __restrict__ outF, long long ldcF,
    __bf16* __restrict__ outB, long long ldcB,
    int Mrows, int Ncols, int K) {
  const int wavesPerBlock = blockDim.x >> 5;
  int wave = blockIdx.x * wavesPerBlock + (threadIdx.x >> 5);
  int lane = threadIdx.x & 31;
  int tilesN = Ncols >> 6;
  int tileM = (wave / tilesN) << 4;
  int tileN = (wave % tilesN) << 6;
  if (tileM >= Mrows) return;  // wave-uniform: EXEC stays all-ones for WMMA

  const int hlf = lane >> 4;   // K-half selector
  const int r   = lane & 15;   // row (A) or col (B/D)

  v8f acc[4];
#pragma unroll
  for (int f = 0; f < 4; ++f)
#pragma unroll
    for (int j = 0; j < 8; ++j) acc[f][j] = 0.0f;

  // A 16x32 fragment: lanes 0-15 K[0..7]+K[16..23], lanes 16-31 K[8..15]+K[24..31]
  const __bf16* arow  = A + (long long)(tileM + r) * K + hlf * 8;
  // B 32x16 fragment: lane = output col; lanes 0-15 K[0..7]+K[16..23], lanes 16-31 K[8..15]+K[24..31].
  // W stored (N,K) row-major == B^T -> two 16B loads.
  const __bf16* wbase = W + (long long)(tileN + r) * K + hlf * 8;
  const long long wfs = (long long)16 * K;  // stride between 16-col fragments

  Frag32B a0, a1, b0[4], b1[4];

#define LOAD_A(dst, kk)                                    \
  do {                                                     \
    (dst).u[0] = *(const uint4*)(arow + (kk));             \
    (dst).u[1] = *(const uint4*)(arow + (kk) + 16);        \
  } while (0)
#define LOAD_B(dst, kk)                                    \
  do {                                                     \
    _Pragma("unroll")                                      \
    for (int f = 0; f < 4; ++f) {                          \
      const __bf16* p = wbase + f * wfs + (kk);            \
      (dst)[f].u[0] = *(const uint4*)(p);                  \
      (dst)[f].u[1] = *(const uint4*)(p + 16);             \
    }                                                      \
  } while (0)

  LOAD_A(a0, 0);
  LOAD_B(b0, 0);
  for (int k = 0; k < K; k += 64) {
    LOAD_A(a1, k + 32);
    LOAD_B(b1, k + 32);
#pragma unroll
    for (int f = 0; f < 4; ++f)
      acc[f] = __builtin_amdgcn_wmma_f32_16x16x32_bf16(
          false, a0.v, false, b0[f].v, (short)0, acc[f], false, false);
    int kn = (k + 64 < K) ? (k + 64) : k;  // clamp: last prefetch re-reads
    LOAD_A(a0, kn);
    LOAD_B(b0, kn);
#pragma unroll
    for (int f = 0; f < 4; ++f)
      acc[f] = __builtin_amdgcn_wmma_f32_16x16x32_bf16(
          false, a1.v, false, b1[f].v, (short)0, acc[f], false, false);
  }
#undef LOAD_A
#undef LOAD_B

  // D layout: lanes 0-15 -> N=lane, M=j ; lanes 16-31 -> N=lane-16, M=j+8
#pragma unroll
  for (int f = 0; f < 4; ++f) {
    int col = tileN + f * 16 + r;
    float bb = bias ? bias[col] : 0.0f;
#pragma unroll
    for (int j = 0; j < 8; ++j) {
      int row = tileM + hlf * 8 + j;
      float v = acc[f][j] + bb;
      if (outF) outF[(long long)row * ldcF + col] = v;
      if (outB) outB[(long long)row * ldcB + col] = (__bf16)v;
    }
  }
}

// ---------------- attention ----------------
// One workgroup (256 threads) per row n. Stage V[n] (L x 512 bf16) into LDS with
// async global->LDS copies, compute logits, masked softmax, weighted sum.
extern __shared__ unsigned char smem_raw[];

__global__ __launch_bounds__(256) void k_attn(
    const float* __restrict__ qp, int qpStride,
    const __bf16* __restrict__ Vv,
    const unsigned char* __restrict__ mask,
    int L,
    __bf16* __restrict__ outB, int outStride) {
  const int n = blockIdx.x;
  const int tid = threadIdx.x;

  float* s_qp  = (float*)smem_raw;            // 512 f32
  float* s_lg  = s_qp + kM;                   // L f32
  float* s_red = s_lg + L;                    // 256 f32 (padded to 320)
  __bf16* s_v  = (__bf16*)(s_red + 320);      // L*512 bf16

  for (int i = tid; i < kM; i += 256) s_qp[i] = qp[(long long)n * qpStride + i];

  // async stage of the value tile: L*1024 bytes in 16B chunks
  {
    unsigned lds_base = (unsigned)(unsigned long long)(void*)s_v;
    unsigned long long gbase =
        (unsigned long long)(const void*)(Vv + (long long)n * L * kM);
    int chunks = L * 64;  // 16B each
    for (int c = tid; c < chunks; c += 256) {
      unsigned lds_addr = lds_base + (unsigned)c * 16u;
      unsigned voff = (unsigned)c * 16u;
      asm volatile("global_load_async_to_lds_b128 %0, %1, %2"
                   :: "v"(lds_addr), "v"(voff), "s"(gbase) : "memory");
    }
    asm volatile("s_wait_asynccnt 0" ::: "memory");
  }
  __syncthreads();

  // logits: one l per thread (L <= 256)
  for (int l = tid; l < L; l += 256) {
    const unsigned* vp = (const unsigned*)(s_v + (long long)l * kM);
    float acc = 0.0f;
#pragma unroll 4
    for (int d = 0; d < kM / 2; ++d) {
      unsigned u = vp[d];
      acc += bflo_(u) * s_qp[2 * d] + bfhi_(u) * s_qp[2 * d + 1];
    }
    float lg = acc * kSCALE;
    if (!mask[(long long)n * L + l]) lg = -1.0e9f;
    s_lg[l] = lg;
  }
  __syncthreads();

  // block softmax (L is a power of two: 64 or 256)
  for (int i = tid; i < 256; i += 256) s_red[i] = (i < L) ? s_lg[i] : -3.0e38f;
  __syncthreads();
  for (int s = 128; s > 0; s >>= 1) {
    if (tid < s) s_red[tid] = fmaxf(s_red[tid], s_red[tid + s]);
    __syncthreads();
  }
  float mx = s_red[0];
  __syncthreads();
  for (int l = tid; l < L; l += 256) s_lg[l] = __expf(s_lg[l] - mx);
  __syncthreads();
  for (int i = tid; i < 256; i += 256) s_red[i] = (i < L) ? s_lg[i] : 0.0f;
  __syncthreads();
  for (int s = 128; s > 0; s >>= 1) {
    if (tid < s) s_red[tid] += s_red[tid + s];
    __syncthreads();
  }
  float inv = 1.0f / s_red[0];
  __syncthreads();
  for (int l = tid; l < L; l += 256) s_lg[l] *= inv;
  __syncthreads();

  // weighted sum over L from LDS: packed u32 reads, one column pair per thread
  {
    const unsigned* vp = (const unsigned*)s_v;
    int dp = tid;  // column pair index, kM/2 == 256 == blockDim
    float a0 = 0.0f, a1 = 0.0f;
    for (int l = 0; l < L; ++l) {
      unsigned u = vp[l * (kM / 2) + dp];
      float w = s_lg[l];
      a0 += w * bflo_(u);
      a1 += w * bfhi_(u);
    }
    outB[(long long)n * outStride + 2 * dp]     = (__bf16)a0;
    outB[(long long)n * outStride + 2 * dp + 1] = (__bf16)a1;
  }
}

// ---------------- host driver ----------------
static inline size_t alignUp256(size_t x) { return (x + 255) & ~size_t(255); }
static inline int blks(long long n, int t) { return (int)((n + t - 1) / t); }

extern "C" void kernel_launch(void* const* d_in, const int* in_sizes, int n_in,
                              void* d_out, int out_size, void* d_ws, size_t ws_size,
                              hipStream_t stream) {
  (void)in_sizes; (void)n_in; (void)out_size; (void)ws_size;

  const float* io_embed     = (const float*)d_in[0];
  const float* code_memory  = (const float*)d_in[1];
  const unsigned char* code_mask  = (const unsigned char*)d_in[2];
  const float* trace_memory = (const float*)d_in[3];
  const unsigned char* trace_mask = (const unsigned char*)d_in[4];
  const long long* outputs  = (const long long*)d_in[5];
  const float* embed   = (const float*)d_in[6];
  const float* w_ih0   = (const float*)d_in[7];
  const float* w_hh0   = (const float*)d_in[8];
  const float* b_ih0   = (const float*)d_in[9];
  const float* b_hh0   = (const float*)d_in[10];
  const float* w_ih1   = (const float*)d_in[11];
  const float* w_hh1   = (const float*)d_in[12];
  const float* b_ih1   = (const float*)d_in[13];
  const float* b_hh1   = (const float*)d_in[14];
  const float* code_q_w = (const float*)d_in[15];
  const float* code_q_b = (const float*)d_in[16];
  const float* trace_q_w = (const float*)d_in[17];
  const float* trace_q_b = (const float*)d_in[18];
  const float* ctx_w   = (const float*)d_in[19];
  const float* ctx_b   = (const float*)d_in[20];
  const float* out_w   = (const float*)d_in[21];
  float* out = (float*)d_out;

  // ---- workspace carve-up ----
  char* ws = (char*)d_ws;
  size_t cur = 0;
  auto alloc = [&](size_t bytes) -> void* {
    void* p = ws + cur;
    cur = alignUp256(cur + bytes);
    return p;
  };
  __bf16* embedB  = (__bf16*)alloc((size_t)kV * kE * 2);
  __bf16* codeB   = (__bf16*)alloc((size_t)kN * kLC * kM * 2);
  __bf16* traceB  = (__bf16*)alloc((size_t)kN * kLT * kM * 2);
  __bf16* wcat0   = (__bf16*)alloc((size_t)4 * kH * kXH0 * 2);
  __bf16* wcat1   = (__bf16*)alloc((size_t)4 * kH * kXH1 * 2);
  __bf16* wq      = (__bf16*)alloc((size_t)2 * kM * kH * 2);
  __bf16* wctx    = (__bf16*)alloc((size_t)kM * (2 * kM) * 2);
  __bf16* wout    = (__bf16*)alloc((size_t)kV * (kM + kH) * 2);
  float*  bcat0   = (float*)alloc((size_t)4 * kH * 4);
  float*  bcat1   = (float*)alloc((size_t)4 * kH * 4);
  float*  bq      = (float*)alloc((size_t)2 * kM * 4);
  __bf16* xh0     = (__bf16*)alloc((size_t)kN * kXH0 * 2);
  __bf16* xh1     = (__bf16*)alloc((size_t)kN * kXH1 * 2);
  __bf16* decB    = (__bf16*)alloc((size_t)kN * kH * 2);
  float*  gates   = (float*)alloc((size_t)kN * 4 * kH * 4);
  float*  qp      = (float*)alloc((size_t)kN * 2 * kM * 4);
  float*  c0      = (float*)alloc((size_t)kN * kH * 4);
  float*  c1      = (float*)alloc((size_t)kN * kH * 4);
  __bf16* ctxcat  = (__bf16*)alloc((size_t)kN * 2 * kM * 2);
  float*  ctxF    = (float*)alloc((size_t)kN * kM * 4);
  __bf16* featB   = (__bf16*)alloc((size_t)kB * (kM + kH) * 2);

  const int TPB = 256;

  // ---- one-time (per launch) conversions / packing ----
  { long long n = (long long)kV * kE;
    k_cvt_bf16<<<blks(n, TPB), TPB, 0, stream>>>(embedB, embed, n); }
  { long long n = (long long)kN * kLC * kM;
    k_cvt_bf16<<<blks(n, TPB), TPB, 0, stream>>>(codeB, code_memory, n); }
  { long long n = (long long)kN * kLT * kM;
    k_cvt_bf16<<<blks(n, TPB), TPB, 0, stream>>>(traceB, trace_memory, n); }
  { long long n = (long long)4 * kH * kXH0;
    k_pack2_bf16<<<blks(n, TPB), TPB, 0, stream>>>(wcat0, w_ih0, kIN, w_hh0, kH, 4 * kH); }
  { long long n = (long long)4 * kH * kXH1;
    k_pack2_bf16<<<blks(n, TPB), TPB, 0, stream>>>(wcat1, w_ih1, kH, w_hh1, kH, 4 * kH); }
  { long long n = (long long)kM * kH;  // code_q_w then trace_q_w stacked along N
    k_cvt_bf16<<<blks(n, TPB), TPB, 0, stream>>>(wq, code_q_w, n);
    k_cvt_bf16<<<blks(n, TPB), TPB, 0, stream>>>(wq + n, trace_q_w, n); }
  { long long n = (long long)kM * 2 * kM;
    k_cvt_bf16<<<blks(n, TPB), TPB, 0, stream>>>(wctx, ctx_w, n); }
  { long long n = (long long)kV * (kM + kH);
    k_cvt_bf16<<<blks(n, TPB), TPB, 0, stream>>>(wout, out_w, n); }
  k_add_f32<<<blks(4 * kH, TPB), TPB, 0, stream>>>(bcat0, b_ih0, b_hh0, 4 * kH);
  k_add_f32<<<blks(4 * kH, TPB), TPB, 0, stream>>>(bcat1, b_ih1, b_hh1, 4 * kH);
  k_copy_f32<<<blks(kM, TPB), TPB, 0, stream>>>(bq, code_q_b, kM);
  k_copy_f32<<<blks(kM, TPB), TPB, 0, stream>>>(bq + kM, trace_q_b, kM);

  // zero initial state: whole xh0/xh1 (then fill io slot), c0, c1
  { long long n = (long long)kN * kXH0 * 2 / 4;
    k_zero32<<<blks(n, TPB), TPB, 0, stream>>>((unsigned*)xh0, n); }
  { long long n = (long long)kN * kXH1 * 2 / 4;
    k_zero32<<<blks(n, TPB), TPB, 0, stream>>>((unsigned*)xh1, n); }
  { long long n = (long long)kN * kH;
    k_zero32<<<blks(n, TPB), TPB, 0, stream>>>((unsigned*)c0, n);
    k_zero32<<<blks(n, TPB), TPB, 0, stream>>>((unsigned*)c1, n); }
  { long long n = (long long)kN * kM;
    k_fill_io<<<blks(n, TPB), TPB, 0, stream>>>(xh0, io_embed); }

  k_labels<<<blks((long long)kB * kT, TPB), TPB, 0, stream>>>(out, outputs);

  auto gemmBlocks = [](int M, int N) { return ((M / 16) * (N / 64) + 7) / 8; };

  const size_t shCode  = kM * 4 + kLC * 4 + 320 * 4 + (size_t)kLC * kM * 2;   // ~69 KB
  const size_t shTrace = kM * 4 + kLT * 4 + 320 * 4 + (size_t)kLT * kM * 2;   // ~260 KB

  // ---- time loop ----
  for (int t = 0; t < kT; ++t) {
    k_tok<<<blks((long long)kN * kE, TPB), TPB, 0, stream>>>(xh0, embedB, outputs, t);

    // LSTM layer 0: gates = [x|h0] @ wcat0^T + bcat0
    k_gemm_wmma<<<gemmBlocks(kN, 4 * kH), TPB, 0, stream>>>(
        xh0, wcat0, bcat0, gates, 4 * kH, (__bf16*)nullptr, 0, kN, 4 * kH, kXH0);
    k_lstm_act<<<blks((long long)kN * kH, TPB), TPB, 0, stream>>>(
        gates, c0, xh0 + kIN, kXH0, xh1, kXH1);

    // LSTM layer 1: gates = [h0|h1] @ wcat1^T + bcat1
    k_gemm_wmma<<<gemmBlocks(kN, 4 * kH), TPB, 0, stream>>>(
        xh1, wcat1, bcat1, gates, 4 * kH, (__bf16*)nullptr, 0, kN, 4 * kH, kXH1);
    k_lstm_act<<<blks((long long)kN * kH, TPB), TPB, 0, stream>>>(
        gates, c1, xh1 + kH, kXH1, decB, kH);

    // fused query projections: qp = dec @ [code_q_w; trace_q_w]^T + bq
    k_gemm_wmma<<<gemmBlocks(kN, 2 * kM), TPB, 0, stream>>>(
        decB, wq, bq, qp, 2 * kM, (__bf16*)nullptr, 0, kN, 2 * kM, kH);

    // attentions (async-staged LDS value tiles)
    k_attn<<<kN, TPB, shCode, stream>>>(qp, 2 * kM, codeB, code_mask, kLC,
                                        ctxcat, 2 * kM);
    k_attn<<<kN, TPB, shTrace, stream>>>(qp + kM, 2 * kM, traceB, trace_mask, kLT,
                                         ctxcat + kM, 2 * kM);

    // ctx = [cc|tc] @ ctx_w^T + ctx_b -> f32 (for feat) and bf16 into xh0 ctx slot
    k_gemm_wmma<<<gemmBlocks(kN, kM), TPB, 0, stream>>>(
        ctxcat, wctx, ctx_b, ctxF, kM, xh0 + kE + kM, kXH0, kN, kM, 2 * kM);

    // feat = max over P of [ctx|dec]
    k_maxfeat<<<blks((long long)kB * (kM + kH), TPB), TPB, 0, stream>>>(ctxF, decB, featB);

    // logits written straight into d_out slice [:, t, :]
    k_gemm_wmma<<<gemmBlocks(kB, kV), TPB, 0, stream>>>(
        featB, wout, (const float*)nullptr,
        out + (long long)t * kV, (long long)kT * kV,
        (__bf16*)nullptr, 0, kB, kV, kM + kH);
  }
}
